// DAPPairformerLayer_36713380446452
// MI455X (gfx1250) — matrix operand order repaired and checked by reading, not compiled, not run on hardware
//
#include <hip/hip_runtime.h>
#include <cstdint>

typedef __bf16 bf16_t;
typedef __attribute__((ext_vector_type(16))) __bf16 bf16x16;
typedef __attribute__((ext_vector_type(8))) float f32x8;

// ---------------------------------------------------------------------------
// Param indices: jax pytree flattening sorts dict keys alphabetically.
// d_in: [0]=mask [1]=pair_mask [2..60]=params(sorted) [61]=s [62]=z
// ---------------------------------------------------------------------------
enum {
  P_att_g = 0, P_att_k, P_att_o, P_att_q, P_att_qb, P_att_v, P_att_z,
  P_att_z_ln_b, P_att_z_ln_g,
  P_pns_b, P_pns_g, P_spn_b, P_spn_g,
  P_tae_b, P_tae_gate, P_tae_k, P_tae_ln_b, P_tae_ln_g, P_tae_o, P_tae_q, P_tae_v,
  P_tas_b, P_tas_gate, P_tas_k, P_tas_ln_b, P_tas_ln_g, P_tas_o, P_tas_q, P_tas_v,
  P_tmi_ag, P_tmi_ap, P_tmi_bg, P_tmi_bp, P_tmi_g, P_tmi_ln_b, P_tmi_ln_g,
  P_tmi_lno_b, P_tmi_lno_g, P_tmi_o,
  P_tmo_ag, P_tmo_ap, P_tmo_bg, P_tmo_bp, P_tmo_g, P_tmo_ln_b, P_tmo_ln_g,
  P_tmo_lno_b, P_tmo_lno_g, P_tmo_o,
  P_ts_ln_b, P_ts_ln_g, P_ts_w1, P_ts_w2, P_ts_w3,
  P_tz_ln_b, P_tz_ln_g, P_tz_w1, P_tz_w2, P_tz_w3
};

__device__ __forceinline__ void put2(bf16x16& v, int i2, unsigned bits) {
  union { unsigned u; bf16_t h[2]; } cv; cv.u = bits;
  v[i2] = cv.h[0]; v[i2 + 1] = cv.h[1];
}
__device__ __forceinline__ float sigf(float x) { return 1.f / (1.f + __expf(-x)); }

// ---------------------------------------------------------------------------
// WMMA GEMM, NT form only: A [M,K] row-major, B [Npad,K] row-major (weights are
// pre-transposed + zero-padded to Npad%64==0 at conversion time; zero rows add
// zero to the accumulator, stores are col-guarded). Block = 8 waves ->
// 32(M) x 64(N) tile, K step 64 (2 WMMA per barrier pair), register
// double-buffered so global loads overlap the WMMAs. Both operand gathers are
// contiguous-pair ds_load_b32 (no sub-word merges). 32-bit index math.
// ---------------------------------------------------------------------------
__global__ __launch_bounds__(256) void gemm_wmma(
    const bf16_t* __restrict__ A, int aBS,
    const bf16_t* __restrict__ Bm, int bBS,
    int M, int Ncols, int K,
    const float* __restrict__ colBias, int act,
    bf16_t* __restrict__ outB, float* __restrict__ outF, int ostr, int oBS,
    float* __restrict__ resid, const bf16_t* __restrict__ gate,
    int transPair)
{
  __shared__ __align__(16) bf16_t As[32 * 64];   // [m 32][k 64]
  __shared__ __align__(16) bf16_t Bs[64 * 64];   // [n 64][k 64]
  const int bz = blockIdx.z;
  A  += bz * aBS;
  Bm += bz * bBS;
  const int m0 = blockIdx.y * 32;
  const int n0 = blockIdx.x * 64;
  const int tid = threadIdx.x;
  const int lane = tid & 31;
  const int w = tid >> 5;
  const int wm = (w >> 2) << 4;   // 0/16
  const int wn = (w & 3) << 4;    // 0..48
  const int hf = lane >> 4;
  const int l16 = lane & 15;

  f32x8 c = {0.f, 0.f, 0.f, 0.f, 0.f, 0.f, 0.f, 0.f};

  // cooperative tile-load geometry
  const int ar = tid >> 3;           // A: 8 bf16 / thread (rows 0..31)
  const int ac = (tid * 8) & 63;
  const int br = tid >> 2;           // B: 16 bf16 / thread (rows 0..63)
  const int bc = (tid * 16) & 63;

  uint4 aR, bR0, bR1;
  auto loadA = [&](int kc) {
    const bf16_t* ga = A + (m0 + ar) * K + kc + ac;
    aR = *(const uint4*)ga;
    __builtin_prefetch(ga + 128, 0, 3);          // near-cache prefetch, 2 chunks ahead
  };
  auto loadB = [&](int kc) {
    const bf16_t* gb = Bm + (n0 + br) * K + kc + bc;
    bR0 = *(const uint4*)gb;
    bR1 = *(const uint4*)(gb + 8);
  };
  auto stage = [&]() {
    *(uint4*)(As + tid * 8) = aR;
    *(uint4*)(Bs + tid * 16)     = bR0;          // [n][k] flattens to tid*16
    *(uint4*)(Bs + tid * 16 + 8) = bR1;
  };
  auto computeTile = [&](int ks) {
    bf16x16 av, bv;
#pragma unroll
    for (int v = 0; v < 8; v++) {    // A: lane = M, k pairs contiguous
      int dk = ks + hf * 16 + 2 * v;
      put2(av, 2 * v, *(const unsigned*)(As + (wm + l16) * 64 + dk));
    }
#pragma unroll
    for (int v = 0; v < 8; v++) {    // B: lane = N, k pairs contiguous
      int k0 = ks + hf * 16 + 2 * v;
      put2(bv, 2 * v, *(const unsigned*)(Bs + (wn + l16) * 64 + k0));
    }
    c = __builtin_amdgcn_wmma_f32_16x16x32_bf16(false, av, false, bv, (short)0, c, false, false);
  };

  loadA(0);
  loadB(0);
  for (int kc = 0; kc < K; kc += 64) {
    stage();
    __syncthreads();
    if (kc + 64 < K) { loadA(kc + 64); loadB(kc + 64); }  // overlap with WMMAs below
    computeTile(0);
    computeTile(32);
    __syncthreads();
  }

  const int col = n0 + wn + l16;
  if (col < Ncols) {
    float cb = colBias ? colBias[col] : 0.f;
#pragma unroll
    for (int v = 0; v < 8; v++) {
      int row = m0 + wm + v + 8 * hf;
      float x = c[v] + cb;
      if (act == 1)      x = sigf(x);
      else if (act == 2) x = x * sigf(x);
      int idx = (row * Ncols + col) * ostr + bz * oBS;
      if (outB)      outB[idx] = (bf16_t)x;
      else if (outF) outF[idx] = x;
      else if (resid) {
        int rr = row;
        if (transPair) rr = ((row & 255) << 8) | (row >> 8);   // pair dim = 256
        float gm = gate ? (float)gate[row * Ncols + col] : 1.f;
        resid[rr * Ncols + col] += gm * x;
      }
    }
  }
}

// ---------------------------------------------------------------------------
// LayerNorm: one wave per row (wave32 shfl reduction). fp32 or bf16 input,
// bf16 or fp32 output; transPair reads pair-transposed rows (pair dim 256).
// ---------------------------------------------------------------------------
__global__ __launch_bounds__(256) void ln_kernel(
    const float* __restrict__ xf, const bf16_t* __restrict__ xb,
    const float* __restrict__ gg, const float* __restrict__ bb,
    bf16_t* __restrict__ outB, float* __restrict__ outF,
    int rows, int C, int transPair)
{
  int row = blockIdx.x * 8 + (threadIdx.x >> 5);
  int lane = threadIdx.x & 31;
  if (row >= rows) return;
  int src = row;
  if (transPair) src = ((row & 255) << 8) | (row >> 8);
  const int sb = src * C;
  float s = 0.f, s2 = 0.f;
  for (int c0 = lane; c0 < C; c0 += 32) {
    float v = xf ? xf[sb + c0] : (float)xb[sb + c0];
    s += v; s2 += v * v;
  }
#pragma unroll
  for (int o = 16; o > 0; o >>= 1) { s += __shfl_xor(s, o, 32); s2 += __shfl_xor(s2, o, 32); }
  float mean = s / C;
  float var  = s2 / C - mean * mean;
  float rs = rsqrtf(var + 1e-5f);
  const int ob = row * C;
  for (int c0 = lane; c0 < C; c0 += 32) {
    float v = xf ? xf[sb + c0] : (float)xb[sb + c0];
    float y = (v - mean) * rs * gg[c0] + bb[c0];
    if (outB) outB[ob + c0] = (bf16_t)y; else outF[ob + c0] = y;
  }
}

// ---------------------------------------------------------------------------
// Attention, TD = head dim (32 triangle / 24 sequence; guards const-folded).
// Block = (outer i, head h, 32-query block). Q/K/V/O addr ((i*N+j)*H+h)*TD+d.
// S in LDS (fp32) -> softmax -> P (bf16) -> P@V. bias/mask always present.
// ---------------------------------------------------------------------------
template <int TD>
__global__ __launch_bounds__(256) void attn_wmma(
    const bf16_t* __restrict__ Q, const bf16_t* __restrict__ Kp, const bf16_t* __restrict__ Vp,
    const float* __restrict__ biasb, int biasH,
    const float* __restrict__ maskp, int mIStr, int mKStr,
    float scale, int N, int H,
    bf16_t* __restrict__ O)
{
  __shared__ float Sm[32][264];
  __shared__ __align__(8) bf16_t Pm[32][256];
  const int i = blockIdx.z, h = blockIdx.y, j0 = blockIdx.x * 32;
  const int tid = threadIdx.x, lane = tid & 31, w = tid >> 5;
  const int hf = lane >> 4, l16 = lane & 15;
  const int rowBase = i * N;
  auto addr = [&](int j, int d) -> int { return ((rowBase + j) * H + h) * TD + d; };
  const float* bp = biasb + h;

  // ---- stage 1: S = Q K^T * scale + bias + mask
  // Each wave: 2 j-tiles x 2 k-tiles; Q fragment loaded once per j-tile.
#pragma unroll
  for (int jt = 0; jt < 2; jt++) {
    bf16x16 av;
    int jj = j0 + jt * 16 + l16;
#pragma unroll
    for (int v = 0; v < 8; v++) {
      int dk = hf * 16 + 2 * v;
      unsigned u = 0;
      if (dk < TD) u = *(const unsigned*)(Q + addr(jj, dk));
      put2(av, 2 * v, u);
    }
    for (int kt = w; kt < 16; kt += 8) {
      bf16x16 bv;
      int kg = kt * 16 + l16;
#pragma unroll
      for (int v = 0; v < 8; v++) {
        int dk = hf * 16 + 2 * v;
        unsigned u = 0;
        if (dk < TD) u = *(const unsigned*)(Kp + addr(kg, dk));
        put2(bv, 2 * v, u);
      }
      f32x8 c = {0.f, 0.f, 0.f, 0.f, 0.f, 0.f, 0.f, 0.f};
      c = __builtin_amdgcn_wmma_f32_16x16x32_bf16(false, av, false, bv, (short)0, c, false, false);
      float mterm = (1.f - maskp[i * mIStr + kg * mKStr]) * (-1e9f);   // hoisted: no v-dep
#pragma unroll
      for (int v = 0; v < 8; v++) {
        int jr = jt * 16 + v + 8 * hf;
        int jg = j0 + jr;
        Sm[jr][kg] = c[v] * scale + bp[(jg * N + kg) * biasH] + mterm;
      }
    }
  }
  __syncthreads();
  // ---- stage 2: softmax (one thread per query row)
  if (tid < 32) {
    float mx = -3.0e38f;
    for (int k2 = 0; k2 < N; k2++) mx = fmaxf(mx, Sm[tid][k2]);
    float sum = 0.f;
    for (int k2 = 0; k2 < N; k2++) { float e = __expf(Sm[tid][k2] - mx); Sm[tid][k2] = e; sum += e; }
    float inv = 1.f / sum;
    for (int k2 = 0; k2 < N; k2++) Pm[tid][k2] = (bf16_t)(Sm[tid][k2] * inv);
  }
  __syncthreads();
  // ---- stage 3: O = P V  (4 tiles: 2 j-blocks x 2 d-blocks; waves 0..3)
  if (w < 4) {
    int jt = w >> 1, d0 = (w & 1) * 16;
    f32x8 c = {0.f, 0.f, 0.f, 0.f, 0.f, 0.f, 0.f, 0.f};
    int dn = d0 + l16;
    for (int kc = 0; kc < N; kc += 32) {
      bf16x16 av, bv;
      int jr = jt * 16 + l16;
#pragma unroll
      for (int v = 0; v < 8; v++) {
        int dk = hf * 16 + 2 * v;
        put2(av, 2 * v, *(const unsigned*)(&Pm[jr][kc + dk]));
      }
#pragma unroll
      for (int v = 0; v < 8; v++) {
        int k0 = kc + hf * 16 + 2 * v;
        bf16_t e0 = (bf16_t)0.f, e1 = (bf16_t)0.f;
        if (dn < TD) { e0 = Vp[addr(k0, dn)]; e1 = Vp[addr(k0 + 1, dn)]; }
        bv[2 * v] = e0; bv[2 * v + 1] = e1;
      }
      c = __builtin_amdgcn_wmma_f32_16x16x32_bf16(false, av, false, bv, (short)0, c, false, false);
    }
    if (dn < TD) {
#pragma unroll
      for (int v = 0; v < 8; v++) {
        int jg = j0 + jt * 16 + v + 8 * hf;
        O[addr(jg, dn)] = (bf16_t)c[v];
      }
    }
  }
}

// dst[h][i][k] = src[(mode? k*256+i : i*256+k)*128 + h]  (N=256, H=128 fixed)
__global__ __launch_bounds__(256) void transpose_hik(
    const bf16_t* __restrict__ src, bf16_t* __restrict__ dst, int mode)
{
  int idx = blockIdx.x * 256 + threadIdx.x;   // 256*256*128 total, grid exact
  int k  = idx & 255;
  int r  = idx >> 8;
  int ii = r & 255;
  int hh = r >> 8;
  int sidx = mode ? ((k << 8) | ii) : ((ii << 8) | k);
  dst[idx] = src[(sidx << 7) + hh];
}

// out = a*b (*pm[idx>>lgC] when pm != null); all bf16 except pm
__global__ __launch_bounds__(256) void pw_mul(
    const bf16_t* __restrict__ a, const bf16_t* __restrict__ b,
    const float* __restrict__ pm, int lgC, int n, bf16_t* __restrict__ out)
{
  int idx = blockIdx.x * 256 + threadIdx.x;
  if (idx >= n) return;
  float x = (float)a[idx] * (float)b[idx];
  if (pm) x *= pm[idx >> lgC];
  out[idx] = (bf16_t)x;
}

__global__ __launch_bounds__(256) void cvt_f32_bf16(
    const float* __restrict__ s, bf16_t* __restrict__ d, int n)
{
  int idx = blockIdx.x * 256 + threadIdx.x;
  if (idx < n) d[idx] = (bf16_t)s[idx];
}

// weight [K,N] f32 -> NT bf16 [Npad,K], rows n>=N zero-filled
__global__ __launch_bounds__(256) void cvt_t_bf16(
    const float* __restrict__ s, bf16_t* __restrict__ d, int K, int N, int total)
{
  int idx = blockIdx.x * 256 + threadIdx.x;
  if (idx >= total) return;
  int n = idx / K, k = idx - n * K;
  d[idx] = (n < N) ? (bf16_t)s[k * N + n] : (bf16_t)0.f;
}

// ---------------------------------------------------------------------------
extern "C" void kernel_launch(void* const* d_in, const int* in_sizes, int n_in,
                              void* d_out, int out_size, void* d_ws, size_t ws_size,
                              hipStream_t stream)
{
  (void)in_sizes; (void)n_in; (void)out_size; (void)ws_size;
  const int N = 256, CZ = 128, TH = 4, TDH = 32, CS = 384, SH = 16, SDH = 24;
  const int NZ = N * N;   // 65536 pair rows

  const float* in_mask = (const float*)d_in[0];
  const float* in_pm   = (const float*)d_in[1];
  auto PR = [&](int i) { return (const float*)d_in[2 + i]; };
  const float* s_in = (const float*)d_in[61];
  const float* z_in = (const float*)d_in[62];
  float* outp = (float*)d_out;

  // ---- workspace carve (all 256B aligned) ~360MB
  char* base = (char*)d_ws;
  size_t off = 0;
  auto alloc = [&](size_t bytes) -> void* {
    void* p = base + off;
    off += (bytes + 255) & ~(size_t)255;
    return p;
  };
  float*  zbuf = (float*) alloc((size_t)NZ * CZ * 4);
  bf16_t* xln  = (bf16_t*)alloc((size_t)NZ * CZ * 2);
  bf16_t* t0   = (bf16_t*)alloc((size_t)NZ * CZ * 2);
  bf16_t* t1   = (bf16_t*)alloc((size_t)NZ * CZ * 2);
  bf16_t* ta   = (bf16_t*)alloc((size_t)NZ * CZ * 2);
  bf16_t* tb   = (bf16_t*)alloc((size_t)NZ * CZ * 2);
  bf16_t* tc   = (bf16_t*)alloc((size_t)NZ * CZ * 2);
  bf16_t* tg   = (bf16_t*)alloc((size_t)NZ * CZ * 2);
  bf16_t* At   = (bf16_t*)alloc((size_t)NZ * CZ * 2);
  bf16_t* Bt   = (bf16_t*)alloc((size_t)NZ * CZ * 2);
  bf16_t* x2   = (bf16_t*)alloc((size_t)NZ * CZ * 2);
  bf16_t* h1   = (bf16_t*)alloc((size_t)NZ * 512 * 2);
  bf16_t* h2   = (bf16_t*)alloc((size_t)NZ * 512 * 2);
  float*  biasb = (float*)alloc((size_t)NZ * 16 * 4);
  float*  sbuf = (float*) alloc((size_t)N * CS * 4);
  bf16_t* sn   = (bf16_t*)alloc((size_t)N * CS * 2);
  bf16_t* sq   = (bf16_t*)alloc((size_t)N * CS * 2);
  bf16_t* sk   = (bf16_t*)alloc((size_t)N * CS * 2);
  bf16_t* sv   = (bf16_t*)alloc((size_t)N * CS * 2);
  bf16_t* sg   = (bf16_t*)alloc((size_t)N * CS * 2);
  bf16_t* so   = (bf16_t*)alloc((size_t)N * CS * 2);
  bf16_t* sgo  = (bf16_t*)alloc((size_t)N * CS * 2);
  bf16_t* sh1  = (bf16_t*)alloc((size_t)N * 4 * CS * 2);
  bf16_t* sh2  = (bf16_t*)alloc((size_t)N * 4 * CS * 2);
  char*   wreg = (char*)  alloc((size_t)16 << 20);

  // ---- weight conversions: f32 [K,N] -> bf16 NT [Npad,K], Npad%64==0
  size_t woff = 0;
  auto WB = [&](int pi, int K, int Ncols) -> bf16_t* {
    int npad = (Ncols + 63) & ~63;
    int total = npad * K;
    bf16_t* dptr = (bf16_t*)(wreg + woff);
    woff += ((size_t)total * 2 + 255) & ~(size_t)255;
    cvt_t_bf16<<<dim3((unsigned)((total + 255) / 256)), dim3(256), 0, stream>>>(PR(pi), dptr, K, Ncols, total);
    return dptr;
  };
  bf16_t* W_tmo_ag = WB(P_tmo_ag, CZ, CZ); bf16_t* W_tmo_ap = WB(P_tmo_ap, CZ, CZ);
  bf16_t* W_tmo_bg = WB(P_tmo_bg, CZ, CZ); bf16_t* W_tmo_bp = WB(P_tmo_bp, CZ, CZ);
  bf16_t* W_tmo_g  = WB(P_tmo_g,  CZ, CZ); bf16_t* W_tmo_o  = WB(P_tmo_o,  CZ, CZ);
  bf16_t* W_tmi_ag = WB(P_tmi_ag, CZ, CZ); bf16_t* W_tmi_ap = WB(P_tmi_ap, CZ, CZ);
  bf16_t* W_tmi_bg = WB(P_tmi_bg, CZ, CZ); bf16_t* W_tmi_bp = WB(P_tmi_bp, CZ, CZ);
  bf16_t* W_tmi_g  = WB(P_tmi_g,  CZ, CZ); bf16_t* W_tmi_o  = WB(P_tmi_o,  CZ, CZ);
  bf16_t* W_tas_q  = WB(P_tas_q, CZ, CZ); bf16_t* W_tas_k = WB(P_tas_k, CZ, CZ);
  bf16_t* W_tas_v  = WB(P_tas_v, CZ, CZ); bf16_t* W_tas_b = WB(P_tas_b, CZ, TH);
  bf16_t* W_tas_g  = WB(P_tas_gate, CZ, CZ); bf16_t* W_tas_o = WB(P_tas_o, CZ, CZ);
  bf16_t* W_tae_q  = WB(P_tae_q, CZ, CZ); bf16_t* W_tae_k = WB(P_tae_k, CZ, CZ);
  bf16_t* W_tae_v  = WB(P_tae_v, CZ, CZ); bf16_t* W_tae_b = WB(P_tae_b, CZ, TH);
  bf16_t* W_tae_g  = WB(P_tae_gate, CZ, CZ); bf16_t* W_tae_o = WB(P_tae_o, CZ, CZ);
  bf16_t* W_tz_w1  = WB(P_tz_w1, CZ, 4 * CZ); bf16_t* W_tz_w2 = WB(P_tz_w2, CZ, 4 * CZ);
  bf16_t* W_tz_w3  = WB(P_tz_w3, 4 * CZ, CZ);
  bf16_t* W_att_z  = WB(P_att_z, CZ, SH);
  bf16_t* W_att_q  = WB(P_att_q, CS, CS); bf16_t* W_att_k = WB(P_att_k, CS, CS);
  bf16_t* W_att_v  = WB(P_att_v, CS, CS); bf16_t* W_att_g = WB(P_att_g, CS, CS);
  bf16_t* W_att_o  = WB(P_att_o, CS, CS);
  bf16_t* W_ts_w1  = WB(P_ts_w1, CS, 4 * CS); bf16_t* W_ts_w2 = WB(P_ts_w2, CS, 4 * CS);
  bf16_t* W_ts_w3  = WB(P_ts_w3, 4 * CS, CS);

  // ---- launch helpers
  auto gemm = [&](const bf16_t* A, int aBS, const bf16_t* B, int bBS,
                  int M, int Nc, int K, const float* cbias, int act,
                  bf16_t* oB, float* oF, int ostr, int oBS,
                  float* resid, const bf16_t* gate, int tp, int batch) {
    dim3 grid((unsigned)((Nc + 63) / 64), (unsigned)(M / 32), (unsigned)batch);
    gemm_wmma<<<grid, dim3(256), 0, stream>>>(A, aBS, B, bBS, M, Nc, K, cbias, act,
                                              oB, oF, ostr, oBS, resid, gate, tp);
  };
  auto lnL = [&](const float* xf, const bf16_t* xb, const float* gg, const float* bb,
                 bf16_t* oB, float* oF, int rows, int C, int tp) {
    ln_kernel<<<dim3((unsigned)(rows / 8)), dim3(256), 0, stream>>>(xf, xb, gg, bb, oB, oF, rows, C, tp);
  };
  auto pw = [&](const bf16_t* a, const bf16_t* b, const float* pm, int lgC, int n, bf16_t* o) {
    pw_mul<<<dim3((unsigned)((n + 255) / 256)), dim3(256), 0, stream>>>(a, b, pm, lgC, n, o);
  };
  auto tpose = [&](const bf16_t* src, bf16_t* dst, int mode) {
    transpose_hik<<<dim3((unsigned)(NZ * CZ / 256)), dim3(256), 0, stream>>>(src, dst, mode);
  };

  // ---- init residual buffers
  hipMemcpyAsync(zbuf, z_in, (size_t)NZ * CZ * 4, hipMemcpyDeviceToDevice, stream);
  hipMemcpyAsync(sbuf, s_in, (size_t)N * CS * 4, hipMemcpyDeviceToDevice, stream);

  // ---- triangle multiplication (outgoing / incoming)
  auto tri_mul = [&](const float* lng, const float* lnb, bf16_t* Wag, bf16_t* Wap,
                     bf16_t* Wbg, bf16_t* Wbp, bf16_t* Wg, const float* lnog, const float* lnob,
                     bf16_t* Wo, int outgoing) {
    lnL(zbuf, nullptr, lng, lnb, xln, nullptr, NZ, CZ, 0);
    gemm(xln, 0, Wag, 0, NZ, CZ, CZ, nullptr, 1, t0, nullptr, 1, 0, nullptr, nullptr, 0, 1);
    gemm(xln, 0, Wap, 0, NZ, CZ, CZ, nullptr, 0, t1, nullptr, 1, 0, nullptr, nullptr, 0, 1);
    pw(t0, t1, in_pm, 7, NZ * CZ, ta);
    gemm(xln, 0, Wbg, 0, NZ, CZ, CZ, nullptr, 1, t0, nullptr, 1, 0, nullptr, nullptr, 0, 1);
    gemm(xln, 0, Wbp, 0, NZ, CZ, CZ, nullptr, 0, t1, nullptr, 1, 0, nullptr, nullptr, 0, 1);
    pw(t0, t1, in_pm, 7, NZ * CZ, tb);
    gemm(xln, 0, Wg, 0, NZ, CZ, CZ, nullptr, 1, tg, nullptr, 1, 0, nullptr, nullptr, 0, 1);
    int mode = outgoing ? 0 : 1;
    tpose(ta, At, mode);
    tpose(tb, Bt, mode);
    // 128 batched NT GEMMs: x2[i,j,h] = sum_k At[h,i,k]*Bt[h,j,k]
    gemm(At, NZ, Bt, NZ, N, N, N, nullptr, 0, x2, nullptr, CZ, 1, nullptr, nullptr, 0, CZ);
    lnL(nullptr, x2, lnog, lnob, t0, nullptr, NZ, CZ, 0);
    gemm(t0, 0, Wo, 0, NZ, CZ, CZ, nullptr, 0, nullptr, nullptr, 1, 0, zbuf, tg, 0, 1);
  };
  tri_mul(PR(P_tmo_ln_g), PR(P_tmo_ln_b), W_tmo_ag, W_tmo_ap, W_tmo_bg, W_tmo_bp, W_tmo_g,
          PR(P_tmo_lno_g), PR(P_tmo_lno_b), W_tmo_o, 1);
  tri_mul(PR(P_tmi_ln_g), PR(P_tmi_ln_b), W_tmi_ag, W_tmi_ap, W_tmi_bg, W_tmi_bp, W_tmi_g,
          PR(P_tmi_lno_g), PR(P_tmi_lno_b), W_tmi_o, 0);

  // ---- triangle attention (start / end)
  auto tri_attn = [&](const float* lng, const float* lnb, bf16_t* Wq, bf16_t* Wk, bf16_t* Wv,
                      bf16_t* Wb, bf16_t* Wgate, bf16_t* Wo, int start) {
    lnL(zbuf, nullptr, lng, lnb, xln, nullptr, NZ, CZ, start ? 0 : 1);
    gemm(xln, 0, Wq, 0, NZ, CZ, CZ, nullptr, 0, ta, nullptr, 1, 0, nullptr, nullptr, 0, 1);
    gemm(xln, 0, Wk, 0, NZ, CZ, CZ, nullptr, 0, tb, nullptr, 1, 0, nullptr, nullptr, 0, 1);
    gemm(xln, 0, Wv, 0, NZ, CZ, CZ, nullptr, 0, tc, nullptr, 1, 0, nullptr, nullptr, 0, 1);
    gemm(xln, 0, Wb, 0, NZ, TH, CZ, nullptr, 0, nullptr, biasb, 1, 0, nullptr, nullptr, 0, 1);
    gemm(xln, 0, Wgate, 0, NZ, CZ, CZ, nullptr, 1, tg, nullptr, 1, 0, nullptr, nullptr, 0, 1);
    int iS = start ? N : 1, kS = start ? 1 : N;   // pm vs pm^T
    attn_wmma<32><<<dim3(N / 32, TH, N), dim3(256), 0, stream>>>(
        ta, tb, tc, biasb, TH, in_pm, iS, kS, 0.17677669529f, N, TH, x2);
    pw(tg, x2, nullptr, 0, NZ * CZ, t0);          // gate * o
    gemm(t0, 0, Wo, 0, NZ, CZ, CZ, nullptr, 0, nullptr, nullptr, 1, 0, zbuf, nullptr,
         start ? 0 : 1, 1);
  };
  tri_attn(PR(P_tas_ln_g), PR(P_tas_ln_b), W_tas_q, W_tas_k, W_tas_v, W_tas_b, W_tas_g, W_tas_o, 1);
  tri_attn(PR(P_tae_ln_g), PR(P_tae_ln_b), W_tae_q, W_tae_k, W_tae_v, W_tae_b, W_tae_g, W_tae_o, 0);

  // ---- transition_z (SwiGLU)
  lnL(zbuf, nullptr, PR(P_tz_ln_g), PR(P_tz_ln_b), xln, nullptr, NZ, CZ, 0);
  gemm(xln, 0, W_tz_w1, 0, NZ, 4 * CZ, CZ, nullptr, 2, h1, nullptr, 1, 0, nullptr, nullptr, 0, 1);
  gemm(xln, 0, W_tz_w2, 0, NZ, 4 * CZ, CZ, nullptr, 0, h2, nullptr, 1, 0, nullptr, nullptr, 0, 1);
  pw(h1, h2, nullptr, 0, NZ * 4 * CZ, h1);
  gemm(h1, 0, W_tz_w3, 0, NZ, CZ, 4 * CZ, nullptr, 0, nullptr, nullptr, 1, 0, zbuf, nullptr, 0, 1);

  // z is final: copy into output (after s block of N*CS floats)
  hipMemcpyAsync(outp + (size_t)N * CS, zbuf, (size_t)NZ * CZ * 4, hipMemcpyDeviceToDevice, stream);

  // ---- sequence attention with pair bias
  lnL(zbuf, nullptr, PR(P_att_z_ln_g), PR(P_att_z_ln_b), xln, nullptr, NZ, CZ, 0);
  gemm(xln, 0, W_att_z, 0, NZ, SH, CZ, nullptr, 0, nullptr, biasb, 1, 0, nullptr, nullptr, 0, 1);
  lnL(sbuf, nullptr, PR(P_pns_g), PR(P_pns_b), sn, nullptr, N, CS, 0);
  gemm(sn, 0, W_att_q, 0, N, CS, CS, PR(P_att_qb), 0, sq, nullptr, 1, 0, nullptr, nullptr, 0, 1);
  gemm(sn, 0, W_att_k, 0, N, CS, CS, nullptr, 0, sk, nullptr, 1, 0, nullptr, nullptr, 0, 1);
  gemm(sn, 0, W_att_v, 0, N, CS, CS, nullptr, 0, sv, nullptr, 1, 0, nullptr, nullptr, 0, 1);
  gemm(sn, 0, W_att_g, 0, N, CS, CS, nullptr, 1, sg, nullptr, 1, 0, nullptr, nullptr, 0, 1);
  attn_wmma<24><<<dim3(N / 32, SH, 1), dim3(256), 0, stream>>>(
      sq, sk, sv, biasb, SH, in_mask, 0, 1, 0.20412414523f, N, SH, so);
  pw(sg, so, nullptr, 0, N * CS, sgo);
  gemm(sgo, 0, W_att_o, 0, N, CS, CS, nullptr, 0, nullptr, nullptr, 1, 0, sbuf, nullptr, 0, 1);

  // ---- transition_s + post norm
  lnL(sbuf, nullptr, PR(P_ts_ln_g), PR(P_ts_ln_b), sn, nullptr, N, CS, 0);
  gemm(sn, 0, W_ts_w1, 0, N, 4 * CS, CS, nullptr, 2, sh1, nullptr, 1, 0, nullptr, nullptr, 0, 1);
  gemm(sn, 0, W_ts_w2, 0, N, 4 * CS, CS, nullptr, 0, sh2, nullptr, 1, 0, nullptr, nullptr, 0, 1);
  pw(sh1, sh2, nullptr, 0, N * 4 * CS, sh1);
  gemm(sh1, 0, W_ts_w3, 0, N, CS, 4 * CS, nullptr, 0, nullptr, nullptr, 1, 0, sbuf, nullptr, 0, 1);
  lnL(sbuf, nullptr, PR(P_spn_g), PR(P_spn_b), nullptr, outp, N, CS, 0);
}
